// SwinLSTMCell_4183298147032
// MI455X (gfx1250) — compile-verified, matrix operand order checked by reading, not run on hardware
//
#include <hip/hip_runtime.h>
#include <hip/hip_bf16.h>
#include <cstdint>

typedef _Float16 f16;
typedef __attribute__((ext_vector_type(16))) _Float16 v16h;
typedef __attribute__((ext_vector_type(8)))  _Float16 h8;
typedef __attribute__((ext_vector_type(8)))  float    v8f;

#define N_WIN   8192            // 8 * 32 * 32 windows
#define N_TOK   131072          // N_WIN * 16 tokens
#define PIXELS  16384           // 128*128
#define HSZ     25165824        // 8*192*128*128 (one output tensor)

// ---- WMMA fragment helpers (ISA 16-bit A 16x32 layout) ---------------------
__device__ __forceinline__ int kmap_i(int lane, int i) {
  // lanes 0-15: K in {0..7, 16..23}; lanes 16-31: K in {8..15, 24..31}
  return ((lane & 16) ? 8 : 0) + (i < 8 ? i : i + 8);
}

__device__ __forceinline__ v16h frag_packed(const f16* tile, int lane) {
  union { v16h v; h8 h[2]; } u;
  const f16* p = tile + lane * 16;
  u.h[0] = *(const h8*)(p);
  u.h[1] = *(const h8*)(p + 8);
  return u.v;
}

// same layout, but sourced from LDS (emits ds_load_b128)
__device__ __forceinline__ v16h frag_packed_s(const f16* tile, int lane) {
  union { v16h v; h8 h[2]; } u;
  const f16* p = tile + lane * 16;
  u.h[0] = *(const h8*)(p);
  u.h[1] = *(const h8*)(p + 8);
  return u.v;
}

__device__ __forceinline__ v16h frag_rows_h(const f16* rowptr, int kbase, int lane) {
  int k0 = kbase + ((lane & 16) ? 8 : 0);
  union { v16h v; h8 h[2]; } u;
  u.h[0] = *(const h8*)(rowptr + k0);
  u.h[1] = *(const h8*)(rowptr + k0 + 16);
  return u.v;
}

__device__ __forceinline__ v8f wmma_f16(v16h a, v16h b, v8f c) {
  return __builtin_amdgcn_wmma_f32_16x16x32_f16(false, a, false, b, (short)0, c,
                                                false, false);
}

__device__ __forceinline__ float sigm(float x) { return 1.f / (1.f + __expf(-x)); }

// ---- K0: pack weight matrix W[N][K] (row-major, f32) into WMMA B tiles -----
// packed[(kt*NT + nt)*512 + lane*16 + i] = (f16) W[nt*16 + lane%16][kt*32 + kmap]
__global__ void k_pack_w(const float* __restrict__ w, f16* __restrict__ dst,
                         int K, int NT) {
  int total = (K / 32) * NT * 512;
  for (int idx = blockIdx.x * blockDim.x + threadIdx.x; idx < total;
       idx += gridDim.x * blockDim.x) {
    int i    = idx & 15;
    int lane = (idx >> 4) & 31;
    int tile = idx >> 9;
    int nt = tile % NT, kt = tile / NT;
    int n = nt * 16 + (lane & 15);
    int k = kt * 32 + kmap_i(lane, i);
    dst[idx] = (f16)w[(size_t)n * K + k];
  }
}

// ---- K1: in_proj GEMM: proj[t][192] = W_in @ concat(x,h) + b_in ------------
// B tiles staged through LDS in chunks of 4 ntiles (8 ksteps -> 32 KB).
__global__ void k_inproj(const float* __restrict__ x, const float* __restrict__ h,
                         const f16* __restrict__ wP, const float* __restrict__ b_in,
                         float* __restrict__ proj) {
  __shared__ f16 sw[32 * 512];                       // [ks][ntL] tiles, 32 KB
  int wave = (blockIdx.x * blockDim.x + threadIdx.x) >> 5;
  int lane = threadIdx.x & 31;
  int win  = wave;                                   // 8192 waves total
  int b    = win >> 10;
  int rem  = win & 1023;
  int wh = rem >> 5, ww = rem & 31;
  int m  = lane & 15;
  int y  = wh * 4 + (m >> 2);
  int xx = ww * 4 + (m & 3);
  size_t pix = (size_t)y * 128 + xx;

  v16h A[8];
  #pragma unroll
  for (int ks = 0; ks < 8; ++ks) {
    union { v16h v; f16 e[16]; } u;
    #pragma unroll
    for (int i = 0; i < 16; ++i) {
      int k = ks * 32 + kmap_i(lane, i);
      float val = (k < 64) ? x[((size_t)b * 64  + k)        * PIXELS + pix]
                           : h[((size_t)b * 192 + (k - 64)) * PIXELS + pix];
      u.e[i] = (f16)val;
    }
    A[ks] = u.v;
  }
  size_t t0 = (size_t)win * 16;
  int mhi = (lane & 16) ? 8 : 0;

  for (int nc = 0; nc < 3; ++nc) {                   // 3 chunks of 4 ntiles
    int nt0 = nc * 4;
    __syncthreads();
    // cooperative stage: 32 tiles * 64 uint4 = 2048 uint4
    for (int q = threadIdx.x; q < 2048; q += 256) {
      int w4 = q & 63, tl = q >> 6;
      int ks = tl >> 2, ntL = tl & 3;
      ((uint4*)sw)[q] = ((const uint4*)wP)[(size_t)(ks * 12 + nt0 + ntL) * 64 + w4];
    }
    __syncthreads();
    #pragma unroll
    for (int ntL = 0; ntL < 4; ++ntL) {
      v8f acc = {};
      #pragma unroll
      for (int ks = 0; ks < 8; ++ks)
        acc = wmma_f16(A[ks], frag_packed_s(sw + (ks * 4 + ntL) * 512, lane), acc);
      int n = (nt0 + ntL) * 16 + (lane & 15);
      float bias = b_in[n];
      #pragma unroll
      for (int r = 0; r < 8; ++r)
        proj[(t0 + r + mhi) * 192 + n] = acc[r] + bias;
    }
  }
}

// ---- K2: LayerNorm over C=192 per token, emit f16 rows ---------------------
__global__ void k_ln(const float* __restrict__ proj, const float* __restrict__ g,
                     const float* __restrict__ bb, f16* __restrict__ xn) {
  int t = blockIdx.x * blockDim.x + threadIdx.x;     // 131072
  const float* row = proj + (size_t)t * 192;
  float s = 0.f, s2 = 0.f;
  for (int c = 0; c < 192; c += 4) {
    float4 v = *(const float4*)(row + c);
    s  += v.x + v.y + v.z + v.w;
    s2 += v.x * v.x + v.y * v.y + v.z * v.z + v.w * v.w;
  }
  float mu = s * (1.f / 192.f);
  float rs = rsqrtf(s2 * (1.f / 192.f) - mu * mu + 1e-5f);
  f16* orow = xn + (size_t)t * 192;
  for (int c = 0; c < 192; ++c)
    orow[c] = (f16)((row[c] - mu) * rs * g[c] + bb[c]);
}

// ---- K3: QKV GEMM: qkv16[t][576] = xn @ W_qkv^T + b_qkv --------------------
// B tiles staged through LDS in chunks of 6 ntiles (6 ksteps -> 36 KB).
__global__ void k_qkv(const f16* __restrict__ xn, const f16* __restrict__ wP,
                      const float* __restrict__ bias, f16* __restrict__ qkv) {
  __shared__ f16 sw[36 * 512];                       // [ks][ntL], 36 KB
  int wave = (blockIdx.x * blockDim.x + threadIdx.x) >> 5;
  int lane = threadIdx.x & 31;
  int win = wave;
  int m = lane & 15, mhi = (lane & 16) ? 8 : 0;
  size_t t0 = (size_t)win * 16;
  const f16* arow = xn + (t0 + m) * 192;
  v16h A[6];
  #pragma unroll
  for (int ks = 0; ks < 6; ++ks) A[ks] = frag_rows_h(arow, ks * 32, lane);

  for (int nc = 0; nc < 6; ++nc) {                   // 6 chunks of 6 ntiles
    int nt0 = nc * 6;
    __syncthreads();
    for (int q = threadIdx.x; q < 2304; q += 256) {  // 36 tiles * 64 uint4
      int w4 = q & 63, tl = q >> 6;
      int ks = tl / 6, ntL = tl % 6;
      ((uint4*)sw)[q] = ((const uint4*)wP)[(size_t)(ks * 36 + nt0 + ntL) * 64 + w4];
    }
    __syncthreads();
    #pragma unroll
    for (int ntL = 0; ntL < 6; ++ntL) {
      v8f acc = {};
      #pragma unroll
      for (int ks = 0; ks < 6; ++ks)
        acc = wmma_f16(A[ks], frag_packed_s(sw + (ks * 6 + ntL) * 512, lane), acc);
      int n = (nt0 + ntL) * 16 + (lane & 15);
      float bi = bias[n];
      #pragma unroll
      for (int r = 0; r < 8; ++r)
        qkv[(t0 + r + mhi) * 576 + n] = (f16)(acc[r] + bi);
    }
  }
}

// ---- K4: windowed attention + pack output as proj-GEMM A fragments ---------
__global__ void k_attn(const f16* __restrict__ qkv, f16* __restrict__ projA) {
  __shared__ float sc[4][256];           // per-wave 16x16 scores
  __shared__ f16   sout[4][16 * 192];    // per-wave 16x192 attention output
  int wv = threadIdx.x >> 5, lane = threadIdx.x & 31;
  int win = blockIdx.x * 4 + wv;
  size_t t0 = (size_t)win * 16;
  const float scale = 0.14433756729740643f;   // 48^-0.5
  int m = lane & 15, mhi = (lane & 16) ? 8 : 0;

  for (int hh = 0; hh < 4; ++hh) {
    int cb = hh * 48;
    // --- q (A) and k^T (B) fragments, K=48 padded to 64 ---
    const f16* qrow = qkv + (t0 + m) * 576 + cb;
    const f16* krow = qkv + (t0 + (lane & 15)) * 576 + 192 + cb;
    v16h qa0 = frag_rows_h(qrow, 0, lane);
    v16h kb0 = frag_rows_h(krow, 0, lane);
    union { v16h v; f16 e[16]; } uq, uk;
    #pragma unroll
    for (int i = 0; i < 16; ++i) {
      int k = 32 + kmap_i(lane, i);
      uq.e[i] = (k < 48) ? qrow[k] : (f16)0.f;
      uk.e[i] = (k < 48) ? krow[k] : (f16)0.f;
    }
    v8f s0 = {};
    s0 = wmma_f16(qa0, kb0, s0);
    s0 = wmma_f16(uq.v, uk.v, s0);
    #pragma unroll
    for (int r = 0; r < 8; ++r)
      sc[wv][(r + mhi) * 16 + (lane & 15)] = s0[r] * scale;
    __syncthreads();
    // --- softmax over 16 cols, one lane per row ---
    if (lane < 16) {
      float mx = -1e30f;
      for (int j = 0; j < 16; ++j) mx = fmaxf(mx, sc[wv][lane * 16 + j]);
      float sum = 0.f;
      for (int j = 0; j < 16; ++j) {
        float e = __expf(sc[wv][lane * 16 + j] - mx);
        sc[wv][lane * 16 + j] = e;
        sum += e;
      }
      float inv = 1.f / sum;
      for (int j = 0; j < 16; ++j) sc[wv][lane * 16 + j] *= inv;
    }
    __syncthreads();
    // --- scores (A, K=16 padded to 32) x v (B) ---
    union { v16h v; f16 e[16]; } sa;
    #pragma unroll
    for (int i = 0; i < 16; ++i) {
      int k = kmap_i(lane, i);
      sa.e[i] = (k < 16) ? (f16)sc[wv][(lane & 15) * 16 + k] : (f16)0.f;
    }
    #pragma unroll
    for (int nt = 0; nt < 3; ++nt) {
      union { v16h v; f16 e[16]; } vb;
      int nch = cb + nt * 16 + (lane & 15);
      #pragma unroll
      for (int i = 0; i < 16; ++i) {
        int k = kmap_i(lane, i);
        vb.e[i] = (k < 16) ? qkv[(t0 + k) * 576 + 384 + nch] : (f16)0.f;
      }
      v8f o = {};
      o = wmma_f16(sa.v, vb.v, o);
      #pragma unroll
      for (int r = 0; r < 8; ++r)
        sout[wv][(r + mhi) * 192 + nch] = (f16)o[r];
    }
    __syncthreads();
  }
  // --- pack as A fragments for the proj GEMM ---
  #pragma unroll
  for (int ks = 0; ks < 6; ++ks) {
    int k0 = ks * 32 + ((lane & 16) ? 8 : 0);
    const f16* srow = &sout[wv][(lane & 15) * 192];
    h8 a = *(const h8*)(srow + k0);
    h8 c = *(const h8*)(srow + k0 + 16);
    f16* dst = projA + ((size_t)(win * 6 + ks) * 32 + lane) * 16;
    *(h8*)(dst)     = a;
    *(h8*)(dst + 8) = c;
  }
}

// ---- K5: proj GEMM: amap[t][192] = attn_out @ W_proj^T + b_proj ------------
// B tiles staged through LDS in chunks of 6 ntiles (36 KB).
__global__ void k_proj(const f16* __restrict__ projA, const f16* __restrict__ wP,
                       const float* __restrict__ bias, float* __restrict__ amap) {
  __shared__ f16 sw[36 * 512];
  int wave = (blockIdx.x * blockDim.x + threadIdx.x) >> 5;
  int lane = threadIdx.x & 31;
  int win = wave, mhi = (lane & 16) ? 8 : 0;
  size_t t0 = (size_t)win * 16;
  v16h A[6];
  #pragma unroll
  for (int ks = 0; ks < 6; ++ks)
    A[ks] = frag_packed(projA + (size_t)(win * 6 + ks) * 512, lane);

  for (int nc = 0; nc < 2; ++nc) {                   // 2 chunks of 6 ntiles
    int nt0 = nc * 6;
    __syncthreads();
    for (int q = threadIdx.x; q < 2304; q += 256) {
      int w4 = q & 63, tl = q >> 6;
      int ks = tl / 6, ntL = tl % 6;
      ((uint4*)sw)[q] = ((const uint4*)wP)[(size_t)(ks * 12 + nt0 + ntL) * 64 + w4];
    }
    __syncthreads();
    #pragma unroll
    for (int ntL = 0; ntL < 6; ++ntL) {
      v8f acc = {};
      #pragma unroll
      for (int ks = 0; ks < 6; ++ks)
        acc = wmma_f16(A[ks], frag_packed_s(sw + (ks * 6 + ntL) * 512, lane), acc);
      int n = (nt0 + ntL) * 16 + (lane & 15);
      float bi = bias[n];
      #pragma unroll
      for (int r = 0; r < 8; ++r)
        amap[(t0 + r + mhi) * 192 + n] = acc[r] + bi;
    }
  }
}

// ---- K6: GroupNorm stats over token-major amap (per b,g) -------------------
__global__ void k_gnstats_tok(const float* __restrict__ amap, float* __restrict__ stats) {
  int bg = blockIdx.x, b = bg >> 3, g = bg & 7;
  int tid = threadIdx.x;
  float s = 0.f, s2 = 0.f;
  for (int t = tid; t < 16384; t += 256) {
    const float* row = amap + ((size_t)(b * 16384 + t)) * 192 + g * 24;
    for (int c = 0; c < 24; c += 4) {
      float4 v = *(const float4*)(row + c);
      s  += v.x + v.y + v.z + v.w;
      s2 += v.x * v.x + v.y * v.y + v.z * v.z + v.w * v.w;
    }
  }
  __shared__ float rs[256], rq[256];
  rs[tid] = s; rq[tid] = s2;
  __syncthreads();
  for (int st = 128; st > 0; st >>= 1) {
    if (tid < st) { rs[tid] += rs[tid + st]; rq[tid] += rq[tid + st]; }
    __syncthreads();
  }
  if (tid == 0) {
    float mu = rs[0] * (1.f / 393216.f);
    float var = rq[0] * (1.f / 393216.f) - mu * mu;
    stats[bg * 2] = mu;
    stats[bg * 2 + 1] = rsqrtf(var + 1e-5f);
  }
}

// ---- K7: apply GN1 and pack gates-GEMM A fragments -------------------------
__global__ void k_gn1_pack(const float* __restrict__ amap, const float* __restrict__ stats,
                           const float* __restrict__ gg, const float* __restrict__ gb,
                           f16* __restrict__ gatesA) {
  int wave = (blockIdx.x * blockDim.x + threadIdx.x) >> 5;
  int lane = threadIdx.x & 31;
  int win = wave, b = win >> 10;
  const float* row = amap + (size_t)(win * 16 + (lane & 15)) * 192;
  for (int ks = 0; ks < 6; ++ks) {
    union { v16h v; f16 e[16]; h8 h[2]; } u;
    #pragma unroll
    for (int i = 0; i < 16; ++i) {
      int k = ks * 32 + kmap_i(lane, i);
      int grp = k / 24;
      float mu = stats[(b * 8 + grp) * 2], rs = stats[(b * 8 + grp) * 2 + 1];
      u.e[i] = (f16)((row[k] - mu) * rs * gg[k] + gb[k]);
    }
    f16* dst = gatesA + ((size_t)(win * 6 + ks) * 32 + lane) * 16;
    *(h8*)(dst)     = u.h[0];
    *(h8*)(dst + 8) = u.h[1];
  }
}

// ---- K8: gates GEMM (i,f,o,g together) fused with LSTM cell math -----------
// per channel-tile j: stage the 24 needed gate tiles (4 gates x 6 ksteps, 24 KB)
__global__ void k_gates_lstm(const f16* __restrict__ gatesA, const f16* __restrict__ wP,
                             const float* __restrict__ bg, const float* __restrict__ cprev,
                             float* __restrict__ out, float* __restrict__ hnraw) {
  __shared__ f16 sw[24 * 512];                       // [ks][gi], 24 KB
  int wave = (blockIdx.x * blockDim.x + threadIdx.x) >> 5;
  int lane = threadIdx.x & 31;
  int win = wave, b = win >> 10;
  int rem = win & 1023, wh = rem >> 5, ww = rem & 31;
  int nlo = lane & 15, mhi = (lane & 16) ? 8 : 0;
  v16h A[6];
  #pragma unroll
  for (int ks = 0; ks < 6; ++ks)
    A[ks] = frag_packed(gatesA + (size_t)(win * 6 + ks) * 512, lane);
  for (int j = 0; j < 12; ++j) {
    __syncthreads();
    for (int q = threadIdx.x; q < 1536; q += 256) {  // 24 tiles * 64 uint4
      int w4 = q & 63, tl = q >> 6;
      int ks = tl >> 2, gi = tl & 3;
      ((uint4*)sw)[q] = ((const uint4*)wP)[(size_t)(ks * 48 + j + gi * 12) * 64 + w4];
    }
    __syncthreads();
    v8f aI = {}, aF = {}, aO = {}, aG = {};
    #pragma unroll
    for (int ks = 0; ks < 6; ++ks) {
      v16h Af = A[ks];
      aI = wmma_f16(Af, frag_packed_s(sw + (ks * 4 + 0) * 512, lane), aI);
      aF = wmma_f16(Af, frag_packed_s(sw + (ks * 4 + 1) * 512, lane), aF);
      aO = wmma_f16(Af, frag_packed_s(sw + (ks * 4 + 2) * 512, lane), aO);
      aG = wmma_f16(Af, frag_packed_s(sw + (ks * 4 + 3) * 512, lane), aG);
    }
    int ch = j * 16 + nlo;
    float bI = bg[ch], bF = bg[192 + ch], bO = bg[384 + ch], bG = bg[576 + ch];
    #pragma unroll
    for (int r = 0; r < 8; ++r) {
      int M = r + mhi;
      int y = wh * 4 + (M >> 2), xx = ww * 4 + (M & 3);
      size_t idx = (((size_t)b * 192 + ch) * 128 + y) * 128 + xx;
      float iv = sigm(aI[r] + bI), fv = sigm(aF[r] + bF);
      float ov = sigm(aO[r] + bO), gv = tanhf(aG[r] + bG);
      float cn = fv * cprev[idx] + iv * gv;
      out[HSZ + idx] = cn;             // cnext (second tuple element)
      hnraw[idx] = ov * tanhf(cn);     // pre-GN hnext, NCHW
    }
  }
}

// ---- K9: GroupNorm stats over NCHW hnraw (fully contiguous per b,g) --------
__global__ void k_gnstats_nchw(const float* __restrict__ hn, float* __restrict__ stats) {
  int bg = blockIdx.x, tid = threadIdx.x;
  const float* base = hn + (size_t)bg * 24 * 16384;
  float s = 0.f, s2 = 0.f;
  for (int i = tid; i < 98304; i += 256) {          // 393216 floats / 4
    float4 v = *(const float4*)(base + (size_t)i * 4);
    s  += v.x + v.y + v.z + v.w;
    s2 += v.x * v.x + v.y * v.y + v.z * v.z + v.w * v.w;
  }
  __shared__ float rs[256], rq[256];
  rs[tid] = s; rq[tid] = s2;
  __syncthreads();
  for (int st = 128; st > 0; st >>= 1) {
    if (tid < st) { rs[tid] += rs[tid + st]; rq[tid] += rq[tid + st]; }
    __syncthreads();
  }
  if (tid == 0) {
    float mu = rs[0] * (1.f / 393216.f);
    float var = rq[0] * (1.f / 393216.f) - mu * mu;
    stats[bg * 2] = mu;
    stats[bg * 2 + 1] = rsqrtf(var + 1e-5f);
  }
}

// ---- K10: apply GN2, write hnext -------------------------------------------
__global__ void k_gn2_apply(const float* __restrict__ hn, const float* __restrict__ stats,
                            const float* __restrict__ gg, const float* __restrict__ gb,
                            float* __restrict__ out) {
  size_t i = ((size_t)blockIdx.x * blockDim.x + threadIdx.x) * 4;
  int ch = (int)((i >> 14) % 192);
  int b  = (int)(i / ((size_t)192 * 16384));
  int grp = ch / 24;
  float mu = stats[(b * 8 + grp) * 2], rs = stats[(b * 8 + grp) * 2 + 1];
  float ga = gg[ch] * rs, be = gb[ch] - mu * ga;
  float4 v = *(const float4*)(hn + i);
  float4 o;
  o.x = v.x * ga + be; o.y = v.y * ga + be;
  o.z = v.z * ga + be; o.w = v.w * ga + be;
  *(float4*)(out + i) = o;
}

// ---------------------------------------------------------------------------
extern "C" void kernel_launch(void* const* d_in, const int* in_sizes, int n_in,
                              void* d_out, int out_size, void* d_ws, size_t ws_size,
                              hipStream_t stream) {
  const float* x       = (const float*)d_in[0];
  const float* h       = (const float*)d_in[1];
  const float* c       = (const float*)d_in[2];
  const float* w_in    = (const float*)d_in[3];
  const float* b_in    = (const float*)d_in[4];
  const float* ln_g    = (const float*)d_in[5];
  const float* ln_b    = (const float*)d_in[6];
  const float* w_qkv   = (const float*)d_in[7];
  const float* b_qkv   = (const float*)d_in[8];
  const float* w_proj  = (const float*)d_in[9];
  const float* b_proj  = (const float*)d_in[10];
  const float* gn_g    = (const float*)d_in[11];
  const float* gn_b    = (const float*)d_in[12];
  const float* w_gates = (const float*)d_in[13];
  const float* b_gates = (const float*)d_in[14];
  float* out = (float*)d_out;

  char* ws = (char*)d_ws;
  size_t off = 0;
  auto carve = [&](size_t bytes) { size_t o = off; off = (off + bytes + 255) & ~(size_t)255; return o; };
  f16*   wInP    = (f16*)  (ws + carve(8 * 12 * 512 * 2));
  f16*   wQkvP   = (f16*)  (ws + carve(6 * 36 * 512 * 2));
  f16*   wProjP  = (f16*)  (ws + carve(6 * 12 * 512 * 2));
  f16*   wGatesP = (f16*)  (ws + carve(6 * 48 * 512 * 2));
  float* stats1  = (float*)(ws + carve(64 * 2 * 4));
  float* stats2  = (float*)(ws + carve(64 * 2 * 4));
  float* bufA    = (float*)(ws + carve((size_t)N_TOK * 192 * 4));   // proj, then amap
  f16*   buf16   = (f16*)  (ws + carve((size_t)N_TOK * 192 * 2));   // xn, then gatesA
  char*  bufBig  =          ws + carve((size_t)N_TOK * 576 * 2);    // qkv16, then hnraw(f32)
  f16*   projA   = (f16*)  (ws + carve((size_t)N_TOK * 192 * 2));
  f16*   qkv16   = (f16*)bufBig;
  float* hnraw   = (float*)bufBig;                                   // reuse after K4

  // K0: pack weights into WMMA B-fragment layout
  k_pack_w<<<192, 256, 0, stream>>>(w_in,    wInP,    256, 12);
  k_pack_w<<<432, 256, 0, stream>>>(w_qkv,   wQkvP,   192, 36);
  k_pack_w<<<144, 256, 0, stream>>>(w_proj,  wProjP,  192, 12);
  k_pack_w<<<576, 256, 0, stream>>>(w_gates, wGatesP, 192, 48);

  // K1..K10
  k_inproj      <<<1024,  256, 0, stream>>>(x, h, wInP, b_in, bufA);
  k_ln          <<<512,   256, 0, stream>>>(bufA, ln_g, ln_b, buf16);
  k_qkv         <<<1024,  256, 0, stream>>>(buf16, wQkvP, b_qkv, qkv16);
  k_attn        <<<2048,  128, 0, stream>>>(qkv16, projA);
  k_proj        <<<1024,  256, 0, stream>>>(projA, wProjP, b_proj, bufA);
  k_gnstats_tok <<<64,    256, 0, stream>>>(bufA, stats1);
  k_gn1_pack    <<<1024,  256, 0, stream>>>(bufA, stats1, gn_g, gn_b, buf16);
  k_gates_lstm  <<<1024,  256, 0, stream>>>(buf16, wGatesP, b_gates, c, out, hnraw);
  k_gnstats_nchw<<<64,    256, 0, stream>>>(hnraw, stats2);
  k_gn2_apply   <<<24576, 256, 0, stream>>>(hnraw, stats2, gn_g, gn_b, out);
  (void)in_sizes; (void)n_in; (void)out_size; (void)ws_size;
}
